// MultiQueryAttention_83451214561501
// MI455X (gfx1250) — matrix operand order repaired and checked by reading, not compiled, not run on hardware
//
#include <hip/hip_runtime.h>
#include <hip/hip_bf16.h>
#include <math.h>

// ---------------- types ----------------
typedef __bf16 bf16_t;
typedef __attribute__((ext_vector_type(16))) __bf16 v16bf;
typedef __attribute__((ext_vector_type(8)))  __bf16 v8bf;
typedef __attribute__((ext_vector_type(8)))  float  v8f;

#define D_  2048
#define H_  16
#define HD_ 128
#define B_  2
#define S_  2048
#define MTOT (B_ * S_)          // 4096
#define NQKV (D_ + 2 * HD_)     // 2304

#if __has_builtin(__builtin_amdgcn_exp2f)
#define FAST_EXP2(x) __builtin_amdgcn_exp2f(x)
#else
#define FAST_EXP2(x) exp2f(x)
#endif
#define LOG2E 1.4426950408889634f

static __device__ inline v8f wmma_bf16(v16bf a, v16bf b, v8f c) {
  // (neg_a, A, neg_b, B, c_mod, C, reuse_a, reuse_b)
  return __builtin_amdgcn_wmma_f32_16x16x32_bf16(false, a, false, b, (short)0, c,
                                                 false, false);
}

// A-fragment (16x32 bf16): lane half=l/16; row = l%16; elems j -> K=half*16+j.
// Two contiguous 16B chunks.
static __device__ inline v16bf load_a_frag(const bf16_t* __restrict__ rowp, int half) {
  v8bf lo = *(const v8bf*)(rowp + half * 16);
  v8bf hi = *(const v8bf*)(rowp + half * 16 + 8);
  return __builtin_shufflevector(lo, hi, 0, 1, 2, 3, 4, 5, 6, 7,
                                 8, 9, 10, 11, 12, 13, 14, 15);
}

// B-fragment (32x16 bf16): lane column N=l%16; elems j -> K = (l/16)*16 + j.
// One contiguous 32B chunk when the source is contract-major per column.
static __device__ inline v16bf load_b_frag(const bf16_t* __restrict__ colp, int half) {
  return *(const v16bf*)(colp + half * 16);
}

// ---------------- conversion kernels ----------------
__global__ void cvt_f32_bf16(const float* __restrict__ src, bf16_t* __restrict__ dst,
                             int n) {
  for (int i = blockIdx.x * blockDim.x + threadIdx.x; i < n;
       i += gridDim.x * blockDim.x)
    dst[i] = (bf16_t)src[i];
}

// src: K x N (row-major fp32); dst: N x K (row-major bf16) == src transposed
__global__ void tcvt_f32_bf16(const float* __restrict__ src, bf16_t* __restrict__ dst,
                              int K, int N) {
  int total = K * N;
  for (int i = blockIdx.x * blockDim.x + threadIdx.x; i < total;
       i += gridDim.x * blockDim.x) {
    int k = i / N, n = i - k * N;
    dst[(size_t)n * K + k] = (bf16_t)src[i];
  }
}

// ---------------- GEMM mainloop: 32x64 output block per wave (2x4 tiles) -----
// 8 WMMAs per k-step vs 6 fragment loads; accumulator interleave keeps the
// WMMA->WMMA same-D RAW hazard 8 instructions apart.
static __device__ inline void gemm_block_2x4(const bf16_t* __restrict__ A,
                                             const bf16_t* __restrict__ Bt,
                                             int lda, int K, int m0, int n0,
                                             int lane, v8f acc[2][4]) {
  const int half = lane >> 4;
  const int lmod = lane & 15;
  const bf16_t* arow0 = A + (size_t)(m0 + lmod) * lda;
  const bf16_t* arow1 = A + (size_t)(m0 + 16 + lmod) * lda;
  const bf16_t* brow0 = Bt + (size_t)(n0 + lmod) * K;
  const bf16_t* brow1 = Bt + (size_t)(n0 + 16 + lmod) * K;
  const bf16_t* brow2 = Bt + (size_t)(n0 + 32 + lmod) * K;
  const bf16_t* brow3 = Bt + (size_t)(n0 + 48 + lmod) * K;

#pragma unroll
  for (int i = 0; i < 2; ++i)
#pragma unroll
    for (int j = 0; j < 4; ++j)
      acc[i][j] = (v8f){0.f, 0.f, 0.f, 0.f, 0.f, 0.f, 0.f, 0.f};

#pragma unroll 2
  for (int k = 0; k < K; k += 32) {
    if ((k & 63) == 0) {  // one near-cache prefetch per 64-element stride
      __builtin_prefetch(arow0 + k + 1024, 0, 3);
      __builtin_prefetch(arow1 + k + 1024, 0, 3);
      __builtin_prefetch(brow0 + k + 1024, 0, 3);
      __builtin_prefetch(brow2 + k + 1024, 0, 3);
    }
    v16bf a0 = load_a_frag(arow0 + k, half);
    v16bf a1 = load_a_frag(arow1 + k, half);
    v16bf b0 = load_b_frag(brow0 + k, half);
    v16bf b1 = load_b_frag(brow1 + k, half);
    v16bf b2 = load_b_frag(brow2 + k, half);
    v16bf b3 = load_b_frag(brow3 + k, half);
    acc[0][0] = wmma_bf16(a0, b0, acc[0][0]);
    acc[1][0] = wmma_bf16(a1, b0, acc[1][0]);
    acc[0][1] = wmma_bf16(a0, b1, acc[0][1]);
    acc[1][1] = wmma_bf16(a1, b1, acc[1][1]);
    acc[0][2] = wmma_bf16(a0, b2, acc[0][2]);
    acc[1][2] = wmma_bf16(a1, b2, acc[1][2]);
    acc[0][3] = wmma_bf16(a0, b3, acc[0][3]);
    acc[1][3] = wmma_bf16(a1, b3, acc[1][3]);
  }
}

// ---------------- QKV projection ----------------
__global__ __launch_bounds__(256) void qkv_gemm(
    const bf16_t* __restrict__ xb, const bf16_t* __restrict__ WqkvT,
    const float* __restrict__ bqkv, bf16_t* __restrict__ qmat,
    bf16_t* __restrict__ kmat, bf16_t* __restrict__ vT) {
  const int lane = threadIdx.x & 31;
  const int wave = threadIdx.x >> 5;
  const int lmod = lane & 15;
  const int mblk = blockIdx.y * 8 + wave;  // 0..127 (32 rows each)
  const int nblk = blockIdx.x;             // 0..35  (64 cols each)
  const int m0 = mblk * 32;
  const int n0 = nblk * 64;

  v8f acc[2][4];
  gemm_block_2x4(xb, WqkvT, D_, D_, m0, n0, lane, acc);

#pragma unroll
  for (int j = 0; j < 4; ++j) {
    const int n_g = n0 + j * 16 + lmod;
    const float bias = bqkv[n_g];
#pragma unroll
    for (int i = 0; i < 2; ++i) {
#pragma unroll
      for (int r = 0; r < 8; ++r) {
        const int m_row = m0 + i * 16 + ((lane < 16) ? r : r + 8);
        const bf16_t bv = (bf16_t)(acc[i][j][r] + bias);
        if (n_g < D_) {
          qmat[(size_t)m_row * D_ + n_g] = bv;                       // q row-major
        } else if (n_g < D_ + HD_) {
          kmat[(size_t)m_row * HD_ + (n_g - D_)] = bv;               // k row-major
        } else {
          const int bb = m_row >> 11, s = m_row & (S_ - 1);
          vT[((size_t)(bb * HD_ + (n_g - D_ - HD_))) * S_ + s] = bv; // v transposed
        }
      }
    }
  }
}

// ---------------- flash attention (one 16-query tile per wave) ----------------
__global__ __launch_bounds__(128) void mqa_attn(
    const bf16_t* __restrict__ qmat, const bf16_t* __restrict__ kmat,
    const bf16_t* __restrict__ vT, bf16_t* __restrict__ attnout) {
  __shared__ __attribute__((aligned(32))) bf16_t pl[4][16][32];

  const int lane = threadIdx.x & 31;
  const int wave = threadIdx.x >> 5;
  const int half = lane >> 4;
  const int lmod = lane & 15;

  const int task = blockIdx.x * 4 + wave;  // 0..4095
  const int b = task >> 11;                // / (H*S/16)
  const int rem = task & 2047;
  const int h = rem >> 7;                  // / (S/16)
  const int q0 = (rem & 127) << 4;

  // preload Q fragments (16 x 128 contract)
  const bf16_t* qrow = qmat + ((size_t)(b * S_ + q0 + lmod)) * D_ + h * HD_;
  v16bf aq[4];
#pragma unroll
  for (int kk = 0; kk < 4; ++kk) aq[kk] = load_a_frag(qrow + kk * 32, half);

  const bf16_t* kbase = kmat + (size_t)b * S_ * HD_;
  const bf16_t* vbase = vT + (size_t)b * HD_ * S_;

  float mrow[8], lrow[8];
  v8f acc[8];
#pragma unroll
  for (int r = 0; r < 8; ++r) { mrow[r] = -1e30f; lrow[r] = 0.f; }
#pragma unroll
  for (int j = 0; j < 8; ++j) acc[j] = (v8f){0.f, 0.f, 0.f, 0.f, 0.f, 0.f, 0.f, 0.f};

  const float sc = 0.088388347648318447f;  // 1/sqrt(128)

  for (int kb = 0; kb < S_; kb += 32) {
    // ---- scores: 16 queries x 32 keys, contract over HD=128 ----
    v8f s0 = {0.f, 0.f, 0.f, 0.f, 0.f, 0.f, 0.f, 0.f};
    v8f s1 = {0.f, 0.f, 0.f, 0.f, 0.f, 0.f, 0.f, 0.f};
#pragma unroll
    for (int kk = 0; kk < 4; ++kk) {
      v16bf bk0 = *(const v16bf*)(kbase + (size_t)(kb + lmod) * HD_ + kk * 32 + half * 16);
      v16bf bk1 = *(const v16bf*)(kbase + (size_t)(kb + 16 + lmod) * HD_ + kk * 32 + half * 16);
      s0 = wmma_bf16(aq[kk], bk0, s0);
      s1 = wmma_bf16(aq[kk], bk1, s1);
    }
    // ---- online softmax ----
    float e[8];
#pragma unroll
    for (int r = 0; r < 8; ++r) {
      s0[r] *= sc; s1[r] *= sc;
      e[r] = fmaxf(s0[r], s1[r]);
    }
#pragma unroll
    for (int r = 0; r < 8; ++r) {
      e[r] = fmaxf(e[r], __shfl_xor(e[r], 1, 32));
      e[r] = fmaxf(e[r], __shfl_xor(e[r], 2, 32));
      e[r] = fmaxf(e[r], __shfl_xor(e[r], 4, 32));
      e[r] = fmaxf(e[r], __shfl_xor(e[r], 8, 32));
    }
    float p0[8], p1[8], tsum[8];
#pragma unroll
    for (int r = 0; r < 8; ++r) {
      const float mnew = fmaxf(mrow[r], e[r]);
      const float corr = FAST_EXP2((mrow[r] - mnew) * LOG2E);
      mrow[r] = mnew;
      p0[r] = FAST_EXP2((s0[r] - mnew) * LOG2E);
      p1[r] = FAST_EXP2((s1[r] - mnew) * LOG2E);
      tsum[r] = p0[r] + p1[r];
      lrow[r] *= corr;
#pragma unroll
      for (int j = 0; j < 8; ++j) acc[j][r] *= corr;
    }
#pragma unroll
    for (int r = 0; r < 8; ++r) {
      tsum[r] += __shfl_xor(tsum[r], 1, 32);
      tsum[r] += __shfl_xor(tsum[r], 2, 32);
      tsum[r] += __shfl_xor(tsum[r], 4, 32);
      tsum[r] += __shfl_xor(tsum[r], 8, 32);
      lrow[r] += tsum[r];
    }
    // ---- P: C-layout -> LDS -> A-fragment re-layout ----
#pragma unroll
    for (int r = 0; r < 8; ++r) {
      const int m = (lane < 16) ? r : r + 8;
      pl[wave][m][lmod] = (bf16_t)p0[r];
      pl[wave][m][16 + lmod] = (bf16_t)p1[r];
    }
    __syncthreads();
    v16bf pf;
    {
      v8bf lo = *(const v8bf*)&pl[wave][lmod][half * 16];
      v8bf hi = *(const v8bf*)&pl[wave][lmod][half * 16 + 8];
      pf = __builtin_shufflevector(lo, hi, 0, 1, 2, 3, 4, 5, 6, 7,
                                   8, 9, 10, 11, 12, 13, 14, 15);
    }
    __syncthreads();
    // ---- O += P @ V (contract over 32 keys) ----
#pragma unroll
    for (int j = 0; j < 8; ++j) {
      v16bf bv = *(const v16bf*)(vbase + (size_t)(j * 16 + lmod) * S_ + kb + half * 16);
      acc[j] = wmma_bf16(pf, bv, acc[j]);
    }
  }

  // ---- epilogue: normalize, write (b, s, h*HD + hd) ----
#pragma unroll
  for (int j = 0; j < 8; ++j) {
#pragma unroll
    for (int r = 0; r < 8; ++r) {
      const int m = (lane < 16) ? r : r + 8;
      const float o = acc[j][r] / lrow[r];
      attnout[((size_t)(b * S_ + q0 + m)) * D_ + h * HD_ + j * 16 + lmod] = (bf16_t)o;
    }
  }
}

// ---------------- output projection ----------------
__global__ __launch_bounds__(256) void out_gemm(
    const bf16_t* __restrict__ attnout, const bf16_t* __restrict__ WoutT,
    const float* __restrict__ bout, float* __restrict__ y) {
  const int lane = threadIdx.x & 31;
  const int wave = threadIdx.x >> 5;
  const int lmod = lane & 15;
  const int mblk = blockIdx.y * 8 + wave;  // 0..127
  const int nblk = blockIdx.x;             // 0..31
  const int m0 = mblk * 32;
  const int n0 = nblk * 64;

  v8f acc[2][4];
  gemm_block_2x4(attnout, WoutT, D_, D_, m0, n0, lane, acc);

#pragma unroll
  for (int j = 0; j < 4; ++j) {
    const int n_g = n0 + j * 16 + lmod;
    const float bias = bout[n_g];
#pragma unroll
    for (int i = 0; i < 2; ++i) {
#pragma unroll
      for (int r = 0; r < 8; ++r) {
        const int m_row = m0 + i * 16 + ((lane < 16) ? r : r + 8);
        y[(size_t)m_row * D_ + n_g] = acc[i][j][r] + bias;
      }
    }
  }
}

// ---------------- host ----------------
extern "C" void kernel_launch(void* const* d_in, const int* in_sizes, int n_in,
                              void* d_out, int out_size, void* d_ws, size_t ws_size,
                              hipStream_t stream) {
  const float* x    = (const float*)d_in[0];  // (B,S,D)
  const float* Wqkv = (const float*)d_in[1];  // (D, D+2*HD)
  const float* bqkv = (const float*)d_in[2];  // (D+2*HD)
  const float* Wout = (const float*)d_in[3];  // (D, D)
  const float* bout = (const float*)d_in[4];  // (D)
  float* y = (float*)d_out;

  char* ws = (char*)d_ws;
  size_t o = 0;
  auto carve = [&](size_t bytes) {
    void* p = ws + o;
    o += (bytes + 255) & ~(size_t)255;
    return p;
  };
  bf16_t* xb      = (bf16_t*)carve((size_t)MTOT * D_ * 2);
  bf16_t* WqkvT   = (bf16_t*)carve((size_t)NQKV * D_ * 2);
  bf16_t* WoutT   = (bf16_t*)carve((size_t)D_ * D_ * 2);
  bf16_t* qmat    = (bf16_t*)carve((size_t)MTOT * D_ * 2);
  bf16_t* kmat    = (bf16_t*)carve((size_t)MTOT * HD_ * 2);
  bf16_t* vT      = (bf16_t*)carve((size_t)B_ * HD_ * S_ * 2);
  bf16_t* attnout = (bf16_t*)carve((size_t)MTOT * D_ * 2);
  (void)ws_size; (void)in_sizes; (void)n_in; (void)out_size;

  // 1) precision conversion (+ weight transposes for contiguous B-fragments)
  cvt_f32_bf16<<<2048, 256, 0, stream>>>(x, xb, MTOT * D_);
  tcvt_f32_bf16<<<2048, 256, 0, stream>>>(Wqkv, WqkvT, D_, NQKV);
  tcvt_f32_bf16<<<2048, 256, 0, stream>>>(Wout, WoutT, D_, D_);

  // 2) QKV projection: 128 x 36 blocks of 32x64, 8 waves/block
  qkv_gemm<<<dim3(NQKV / 64, MTOT / 32 / 8), 256, 0, stream>>>(
      xb, WqkvT, bqkv, qmat, kmat, vT);

  // 3) flash attention: 4096 query tiles, 4 waves/block
  mqa_attn<<<dim3((B_ * H_ * (S_ / 16)) / 4), 128, 0, stream>>>(
      qmat, kmat, vT, attnout);

  // 4) output projection: 128 x 32 blocks of 32x64
  out_gemm<<<dim3(D_ / 64, MTOT / 32 / 8), 256, 0, stream>>>(
      attnout, WoutT, bout, y);
}
